// ProjectedGradientDescent_42417097015890
// MI455X (gfx1250) — compile-verified, matrix operand order checked by reading, not run on hardware
//
#include <hip/hip_runtime.h>
#include <math.h>
#include <stdint.h>

// PGD on density matrices, eigh replaced by GEMM-only PSD projection:
//   P = 0.5*(M + c * (M/c) @ sign(M/c)),  c = ||M||_F,
//   sign via Newton-Schulz X <- 1.5X - 0.5 X^3 (all bf16 WMMA GEMMs).
// All operands are symmetric -> B-operand "columns" are rows -> all tile
// loads are contiguous row-major b128; no transpose path needed.
// GEMM inner loop uses CDNA5 async global->LDS copies (ASYNCcnt) with
// double-buffered LDS tiles (rows padded to 80 bf16 to kill LDS bank
// conflicts on the WMMA fragment loads).

#define PGD_LR   0.1f
#define PGD_STEPS 10
#define NS_ITERS  14

#define LDW 80  // padded LDS row stride in bf16 (160 B: 32B-aligned, ~conflict-free)

typedef __bf16 bf16_t;
typedef __attribute__((ext_vector_type(16))) __bf16 v16bf;
typedef __attribute__((ext_vector_type(8)))  float  v8f;

// ---------------------------------------------------------------- scalars
__global__ void zero_scalars_kernel(float* sc) {
  if (threadIdx.x < 3) sc[threadIdx.x] = 0.0f;
}

// sum((a-b)^2) if b != null, else sum(a^2), accumulated into *acc
__global__ __launch_bounds__(256) void reduce_sq_kernel(
    const float* __restrict__ a, const float* __restrict__ b,
    float* __restrict__ acc, size_t n) {
  __shared__ float sdata[256];
  const int tid = threadIdx.x;
  float s = 0.0f;
  for (size_t i = (size_t)blockIdx.x * 256 + tid; i < n;
       i += (size_t)gridDim.x * 256) {
    float d = b ? (a[i] - b[i]) : a[i];
    s += d * d;
  }
  sdata[tid] = s;
  __syncthreads();
  for (int st = 128; st > 0; st >>= 1) {
    if (tid < st) sdata[tid] += sdata[tid + st];
    __syncthreads();
  }
  if (tid == 0) atomicAdd(acc, sdata[0]);
}

__global__ __launch_bounds__(256) void trace_kernel(
    const float* __restrict__ m, float* __restrict__ acc, int N) {
  __shared__ float sdata[256];
  const int tid = threadIdx.x;
  float s = 0.0f;
  for (int i = blockIdx.x * 256 + tid; i < N; i += gridDim.x * 256)
    s += m[(size_t)i * N + i];
  sdata[tid] = s;
  __syncthreads();
  for (int st = 128; st > 0; st >>= 1) {
    if (tid < st) sdata[tid] += sdata[tid + st];
    __syncthreads();
  }
  if (tid == 0) atomicAdd(acc, sdata[0]);
}

// M = sym(rho - LR * (rho - tgt)/||rho - tgt||_F)   (subgradient 0 at 0)
__global__ __launch_bounds__(256) void grad_step_sym_kernel(
    const float* __restrict__ rho, const float* __restrict__ tgt,
    const float* __restrict__ sc, float* __restrict__ M, int N) {
  size_t idx = (size_t)blockIdx.x * 256 + threadIdx.x;
  size_t NN = (size_t)N * N;
  if (idx >= NN) return;
  int i = (int)(idx / N), j = (int)(idx % N);
  float sq = sc[0];
  float s = (sq > 0.0f) ? (PGD_LR * rsqrtf(sq)) : 0.0f;
  size_t ij = (size_t)i * N + j, ji = (size_t)j * N + i;
  float pij = rho[ij] - s * (rho[ij] - tgt[ij]);
  float pji = rho[ji] - s * (rho[ji] - tgt[ji]);
  M[idx] = 0.5f * (pij + pji);
}

// A = X0 = bf16(M / ||M||_F)
__global__ __launch_bounds__(256) void make_bf16_kernel(
    const float* __restrict__ M, const float* __restrict__ sc,
    bf16_t* __restrict__ A, bf16_t* __restrict__ X, size_t n) {
  size_t i = (size_t)blockIdx.x * 256 + threadIdx.x;
  if (i >= n) return;
  float fro = sc[1];
  float inv = (fro > 0.0f) ? rsqrtf(fro) : 0.0f;
  bf16_t h = (bf16_t)(M[i] * inv);
  A[i] = h;
  X[i] = h;
}

// rho = P / trace(P)
__global__ __launch_bounds__(256) void final_scale_kernel(
    const float* __restrict__ P, const float* __restrict__ sc,
    float* __restrict__ rho, size_t n) {
  size_t i = (size_t)blockIdx.x * 256 + threadIdx.x;
  if (i >= n) return;
  float tr = sc[2];
  float inv = (tr != 0.0f) ? (1.0f / tr) : 0.0f;
  rho[i] = P[i] * inv;
}

// ---------------------------------------------------------------- GEMM
// Async 16-byte copy global -> LDS, tracked by ASYNCcnt. The flat-LDS
// aperture keeps the LDS byte offset in addr[31:0], so truncating the
// generic pointer yields the LDS destination address.
__device__ __forceinline__ void async_b128(const bf16_t* g, uint32_t lds) {
  asm volatile("global_load_async_to_lds_b128 %0, %1, off"
               :
               : "v"(lds), "v"((const void*)g)
               : "memory");
}

// D = A @ B (4096^3, bf16 in / f32 acc) with fused epilogue:
//   MODE 1: outH = bf16(D)                          (T = X*X)
//   MODE 2: outH = bf16(1.5*auxH - 0.5*D)           (NS update, auxH = X)
//   MODE 3: auxF = 0.5*(auxF + c*D), c=sqrt(sc[1])  (P = 0.5(M + |M|))
// Block = 8 wave32 -> 64x128 tile; each wave 32x32 via 2x2 WMMA frags.
// BK=64 K-chunks, double-buffered padded LDS (60 KB) fed by async copies.
template <int MODE>
__global__ __launch_bounds__(256) void gemm_ns(
    const bf16_t* __restrict__ A, const bf16_t* __restrict__ B,
    bf16_t* __restrict__ outH, const bf16_t* __restrict__ auxH,
    float* __restrict__ auxF, const float* __restrict__ sc, int N) {
  __shared__ __align__(32) bf16_t Asm[2][64 * LDW];    // 64 rows x BK (padded)
  __shared__ __align__(32) bf16_t Bsm[2][128 * LDW];   // 128 "cols" x BK

  const int tid  = threadIdx.x;
  const int m0   = blockIdx.y * 64;
  const int n0   = blockIdx.x * 128;
  const int wave = tid >> 5;
  const int lane = tid & 31;
  const int wm   = (wave >> 2) * 32;  // 0 / 32
  const int wn   = (wave & 3) * 32;   // 0 / 32 / 64 / 96
  const int lr   = lane & 15;
  const int kh   = lane >> 4;

  // Loader mapping: BK=64 -> 8 uint4 per row; 256 threads.
  // A: 64 rows * 8 = 512 xfers -> 2/thread (rows r, r+32).
  // B: 128 rows * 8 = 1024 xfers -> 4/thread (rows r, r+32, r+64, r+96).
  const int lrow = tid >> 3;            // 0..31
  const int lseg = (tid & 7) * 8;       // bf16 element offset within row

  const bf16_t* gA0 = A + (size_t)(m0 + lrow) * N + lseg;
  const bf16_t* gA1 = A + (size_t)(m0 + lrow + 32) * N + lseg;
  const bf16_t* gB[4];
#pragma unroll
  for (int i = 0; i < 4; ++i)
    gB[i] = B + (size_t)(n0 + lrow + 32 * i) * N + lseg;

  uint32_t lA0[2], lA1[2], lB[2][4];
#pragma unroll
  for (int b = 0; b < 2; ++b) {
    lA0[b] = (uint32_t)(uintptr_t)(&Asm[b][lrow * LDW + lseg]);
    lA1[b] = (uint32_t)(uintptr_t)(&Asm[b][(lrow + 32) * LDW + lseg]);
#pragma unroll
    for (int i = 0; i < 4; ++i)
      lB[b][i] = (uint32_t)(uintptr_t)(&Bsm[b][(lrow + 32 * i) * LDW + lseg]);
  }

  v8f acc[2][2] = {};

  // Issue one chunk's 6 async transfers (per lane) and advance along K.
  auto issue = [&](int b) {
    async_b128(gA0, lA0[b]);
    async_b128(gA1, lA1[b]);
#pragma unroll
    for (int i = 0; i < 4; ++i) async_b128(gB[i], lB[b][i]);
    gA0 += 64;
    gA1 += 64;
#pragma unroll
    for (int i = 0; i < 4; ++i) gB[i] += 64;
  };

  // 8 WMMAs on one resident buffer.
  auto compute = [&](int b) {
#pragma unroll
    for (int ks = 0; ks < 64; ks += 32) {
      // Fragments in contiguous k order (same for A and B): lanes 0-15 -> K 0..15, lanes 16-31 -> 16..31
      v16bf af0 = *(const v16bf*)(&Asm[b][(wm + lr) * LDW + ks + kh * 16]);
      v16bf af1 = *(const v16bf*)(&Asm[b][(wm + 16 + lr) * LDW + ks + kh * 16]);
      v16bf bf0 = *(const v16bf*)(&Bsm[b][(wn + lr) * LDW + ks + kh * 16]);
      v16bf bf1 = *(const v16bf*)(&Bsm[b][(wn + 16 + lr) * LDW + ks + kh * 16]);
      acc[0][0] = __builtin_amdgcn_wmma_f32_16x16x32_bf16(
          false, af0, false, bf0, (short)0, acc[0][0], false, false);
      acc[0][1] = __builtin_amdgcn_wmma_f32_16x16x32_bf16(
          false, af0, false, bf1, (short)0, acc[0][1], false, false);
      acc[1][0] = __builtin_amdgcn_wmma_f32_16x16x32_bf16(
          false, af1, false, bf0, (short)0, acc[1][0], false, false);
      acc[1][1] = __builtin_amdgcn_wmma_f32_16x16x32_bf16(
          false, af1, false, bf1, (short)0, acc[1][1], false, false);
    }
  };

  const int KC = N / 64;
  int buf = 0;
  issue(0);

  // Steady state: next chunk's copies in flight while this chunk computes.
  for (int k = 0; k < KC - 1; ++k) {
    issue(buf ^ 1);
    asm volatile("s_wait_asynccnt 0x6" ::: "memory");
    __syncthreads();
    compute(buf);
    __syncthreads();  // all waves done reading buf before it is refilled
    buf ^= 1;
  }
  // Peeled last chunk: drain and compute, no trailing barrier needed.
  asm volatile("s_wait_asynccnt 0x0" ::: "memory");
  __syncthreads();
  compute(buf);

  // Epilogue. C/D layout: VGPR r -> M = r + 8*kh, N = lane&15.
  float cval = 0.0f;
  if (MODE == 3) cval = sqrtf(fmaxf(sc[1], 0.0f));

#pragma unroll
  for (int i = 0; i < 2; ++i) {
#pragma unroll
    for (int j = 0; j < 2; ++j) {
      int mb = m0 + wm + i * 16 + 8 * kh;
      int nb = n0 + wn + j * 16 + lr;
#pragma unroll
      for (int r = 0; r < 8; ++r) {
        size_t off = (size_t)(mb + r) * N + nb;
        float d = acc[i][j][r];
        if (MODE == 1) {
          outH[off] = (bf16_t)d;
        } else if (MODE == 2) {
          outH[off] = (bf16_t)(1.5f * (float)auxH[off] - 0.5f * d);
        } else {
          auxF[off] = 0.5f * (auxF[off] + cval * d);
        }
      }
    }
  }
}

// ---------------------------------------------------------------- host
extern "C" void kernel_launch(void* const* d_in, const int* in_sizes, int n_in,
                              void* d_out, int out_size, void* d_ws,
                              size_t ws_size, hipStream_t stream) {
  const float* target = (const float*)d_in[0];
  float* rho = (float*)d_out;

  size_t nn = (size_t)in_sizes[0];
  int N = 1;
  while ((size_t)N * N < nn) N <<= 1;  // 4096
  size_t NN = (size_t)N * N;

  // Workspace layout: [3 scalars | M f32 | A bf16 | X bf16 | Xn bf16 | T bf16]
  char* w = (char*)d_ws;
  float*  sc  = (float*)w;                              // sq, fro, trace
  float*  M   = (float*)(w + 256);                      // NN * 4 B
  bf16_t* Abf = (bf16_t*)(w + 256 + NN * sizeof(float));
  bf16_t* X   = Abf + NN;
  bf16_t* Xn  = X + NN;
  bf16_t* T   = Xn + NN;

  dim3 gemmGrid(N / 128, N / 64);
  int ewBlocks = (int)((NN + 255) / 256);

  for (int step = 0; step < PGD_STEPS; ++step) {
    const float* rin = (step == 0) ? target : (const float*)rho;

    zero_scalars_kernel<<<1, 32, 0, stream>>>(sc);
    reduce_sq_kernel<<<4096, 256, 0, stream>>>(rin, target, sc + 0, NN);
    grad_step_sym_kernel<<<ewBlocks, 256, 0, stream>>>(rin, target, sc, M, N);
    reduce_sq_kernel<<<4096, 256, 0, stream>>>(M, nullptr, sc + 1, NN);
    make_bf16_kernel<<<ewBlocks, 256, 0, stream>>>(M, sc, Abf, X, NN);

    bf16_t* xc = X;
    bf16_t* xn = Xn;
    for (int it = 0; it < NS_ITERS; ++it) {
      // T = bf16(X @ X)
      gemm_ns<1><<<gemmGrid, 256, 0, stream>>>(xc, xc, T, nullptr, nullptr,
                                               sc, N);
      // Xn = bf16(1.5*X - 0.5 * X @ T)
      gemm_ns<2><<<gemmGrid, 256, 0, stream>>>(xc, T, xn, xc, nullptr, sc, N);
      bf16_t* tmp = xc; xc = xn; xn = tmp;
    }

    // M <- P = 0.5*(M + c * A @ sign)   (in place)
    gemm_ns<3><<<gemmGrid, 256, 0, stream>>>(Abf, xc, nullptr, nullptr, M,
                                             sc, N);
    trace_kernel<<<16, 256, 0, stream>>>(M, sc + 2, N);
    final_scale_kernel<<<ewBlocks, 256, 0, stream>>>(M, sc, rho, NN);
  }
}
